// ScannedRNN_81243601371425
// MI455X (gfx1250) — matrix operand order repaired and checked by reading, not compiled, not run on hardware
//
#include <hip/hip_runtime.h>

// ---------------------------------------------------------------------------
// Scanned GRU (flax GRUCell with reset-masked carry) for MI455X (gfx1250).
//
//  P0 pack_weights : Wi,Wh (f32 [H,3H]) -> column-major bf16 [3H][H] so WMMA
//                    B-fragments are two contiguous 16B runs per lane.
//  P1 gi_gemm      : gi = X@Wi + bi over all B*T rows, bf16 WMMA, f32 acc,
//                    full-chip parallel; A/B fragments prefetched one k-step
//                    ahead, with sched_barrier(0) fences so the scheduler
//                    cannot collapse the pipeline (keeps loads 6 WMMAs ahead).
//  P2 gru_scan     : 16 blocks (one per 16-batch-row tile), 8 waves each;
//                    wave w owns h-columns [32w,32w+32). Per step: prefetch
//                    gi gather pre-barrier, mask h by resets, publish bf16 h
//                    tile to LDS, 48 WMMAs with double-buffered Wh fragments
//                    (L2-resident, 384 KB), gate math, carry stays in regs.
// ---------------------------------------------------------------------------

typedef __attribute__((ext_vector_type(16))) __bf16 v16bf;
typedef __attribute__((ext_vector_type(8)))  __bf16 v8bf;
typedef __attribute__((ext_vector_type(8)))  float  v8f;
typedef __attribute__((ext_vector_type(4)))  float  v4f;

#define BATCH 256
#define TT    512
#define HH    256
#define H3    768

union FragU { v16bf v; v8bf h[2]; };
struct ARaw { v4f q[4]; };   // 16 f32 of one A fragment, pre-conversion

// Raw A fragment loads (no conversion -> safe to prefetch without stalling).
// Lane L<16: row L, k = {kt*32+0..7, kt*32+16..23}; lane L+16: {+8..15,+24..31}.
__device__ __forceinline__ ARaw load_a_raw(const float* __restrict__ rowPtr,
                                           int kt, int laneHi) {
  int k0 = kt * 32 + (laneHi ? 8 : 0);
  ARaw r;
  r.q[0] = *(const v4f*)(rowPtr + k0);
  r.q[1] = *(const v4f*)(rowPtr + k0 + 4);
  r.q[2] = *(const v4f*)(rowPtr + k0 + 16);
  r.q[3] = *(const v4f*)(rowPtr + k0 + 20);
  return r;
}

__device__ __forceinline__ v16bf cvt_a(const ARaw& r) {
  FragU u;
#pragma unroll
  for (int q = 0; q < 4; ++q) {
    u.v[q * 4 + 0] = (__bf16)r.q[q][0];
    u.v[q * 4 + 1] = (__bf16)r.q[q][1];
    u.v[q * 4 + 2] = (__bf16)r.q[q][2];
    u.v[q * 4 + 3] = (__bf16)r.q[q][3];
  }
  return u.v;
}

// A fragment from a bf16 LDS tile (row stride HH elements).
__device__ __forceinline__ v16bf load_a_lds(const __bf16* rowPtr, int kt, int laneHi) {
  int k0 = kt * 32 + (laneHi ? 8 : 0);
  FragU u;
  u.h[0] = *(const v8bf*)(rowPtr + k0);
  u.h[1] = *(const v8bf*)(rowPtr + k0 + 16);
  return u.v;
}

// B fragment (32x16 bf16) from packed column-major bf16 weights WT[3H][H]:
// lane holds column nCol+(lane&15), k = {+0..7,+16..23} (lanes<16) / {+8..15,+24..31}.
__device__ __forceinline__ v16bf load_b(const __bf16* __restrict__ WT,
                                        int nCol, int kt, int lane) {
  int n  = nCol + (lane & 15);
  int k0 = kt * 32 + ((lane & 16) ? 8 : 0);
  const __bf16* p = WT + (size_t)n * HH + k0;
  FragU u;
  u.h[0] = *(const v8bf*)p;
  u.h[1] = *(const v8bf*)(p + 16);
  return u.v;
}

__device__ __forceinline__ float sigmoid_f(float v) {
  return 1.0f / (1.0f + __expf(-v));
}
__device__ __forceinline__ float tanh_f(float v) {
  float e = __expf(2.0f * v);
  return (e - 1.0f) / (e + 1.0f);
}

// --------------------------- P0: weight packing ----------------------------
__global__ __launch_bounds__(256) void pack_weights_kernel(
    const float* __restrict__ Wi, const float* __restrict__ Wh,
    __bf16* __restrict__ WiT, __bf16* __restrict__ WhT) {
  int n = blockIdx.x;    // 0..767
  int k = threadIdx.x;   // 0..255
  if (blockIdx.y == 0)
    WiT[(size_t)n * HH + k] = (__bf16)Wi[(size_t)k * H3 + n];
  else
    WhT[(size_t)n * HH + k] = (__bf16)Wh[(size_t)k * H3 + n];
}

// ------------------------- P1: gi = X @ Wi + bi ----------------------------
__global__ __launch_bounds__(256) void gi_gemm_kernel(
    const float* __restrict__ x, const __bf16* __restrict__ WiT,
    const float* __restrict__ bi, float* __restrict__ gi) {
  const int lane    = threadIdx.x & 31;
  const int wave    = threadIdx.x >> 5;
  const int laneHi  = lane >> 4;
  const int laneLo  = lane & 15;
  const int row0    = blockIdx.x * 16;
  const int colBase = wave * 96;

  const float* aRow = x + (size_t)(row0 + laneLo) * HH;

  v8f acc[6];
#pragma unroll
  for (int j = 0; j < 6; ++j) {
    float bv = bi[colBase + j * 16 + laneLo];
#pragma unroll
    for (int i = 0; i < 8; ++i) acc[j][i] = bv;
  }

  // Software-pipelined K loop: prefetch kt+1 A (raw f32) and B fragments
  // while the 6 WMMAs of kt execute. sched_barrier(0) fences keep the
  // prefetch batch ABOVE the WMMAs so the scheduler cannot serialize it.
  ARaw  abuf[2];
  v16bf bbuf[2][6];
  abuf[0] = load_a_raw(aRow, 0, laneHi);
#pragma unroll
  for (int j = 0; j < 6; ++j) bbuf[0][j] = load_b(WiT, colBase + j * 16, 0, lane);

#pragma unroll
  for (int kt = 0; kt < 8; ++kt) {
    const int cur = kt & 1, nxt = cur ^ 1;
    if (kt < 7) {
      abuf[nxt] = load_a_raw(aRow, kt + 1, laneHi);
#pragma unroll
      for (int j = 0; j < 6; ++j)
        bbuf[nxt][j] = load_b(WiT, colBase + j * 16, kt + 1, lane);
    }
    __builtin_amdgcn_sched_barrier(0);  // loads for kt+1 stay above kt's WMMAs
    v16bf a = cvt_a(abuf[cur]);
#pragma unroll
    for (int j = 0; j < 6; ++j)
      acc[j] = __builtin_amdgcn_wmma_f32_16x16x32_bf16(
          false, a, false, bbuf[cur][j], (short)0, acc[j], false, false);
    __builtin_amdgcn_sched_barrier(0);
  }

  // C layout: VGPR i -> row i (lanes 0-15) / row i+8 (lanes 16-31), col=lane&15
#pragma unroll
  for (int j = 0; j < 6; ++j) {
    int col = colBase + j * 16 + laneLo;
#pragma unroll
    for (int i = 0; i < 8; ++i) {
      int r = row0 + i + laneHi * 8;
      gi[(size_t)r * H3 + col] = acc[j][i];
    }
  }
}

// ------------------------------ P2: the scan -------------------------------
__global__ __launch_bounds__(256) void gru_scan_kernel(
    const float* __restrict__ x, const int* __restrict__ resets,
    const __bf16* __restrict__ WiT, const __bf16* __restrict__ WhT,
    const float* __restrict__ bi, const float* __restrict__ bn,
    const float* __restrict__ h0, const float* __restrict__ gi,  // may be null
    float* __restrict__ out) {
  __shared__ __align__(16) __bf16 h_bf[16 * HH];
  __shared__ __align__(16) __bf16 x_bf[16 * HH];  // fused path only

  const int lane   = threadIdx.x & 31;
  const int wave   = threadIdx.x >> 5;
  const int laneHi = lane >> 4;
  const int laneLo = lane & 15;
  const int bb     = blockIdx.x * 16;  // batch tile base
  const int cR0    = wave * 32;        // this wave's h-column base
  const bool fused = (gi == nullptr);

  // carry in registers; lane (p,i) <-> (row = i + 8*laneHi, col = cR0+16p+laneLo)
  float hreg[2][8];
#pragma unroll
  for (int p = 0; p < 2; ++p) {
    int col = cR0 + p * 16 + laneLo;
#pragma unroll
    for (int i = 0; i < 8; ++i)
      hreg[p][i] = h0[(size_t)(bb + i + laneHi * 8) * HH + col];
  }

  float bnv[2];
#pragma unroll
  for (int p = 0; p < 2; ++p) bnv[p] = bn[cR0 + p * 16 + laneLo];
  float biv[6];
#pragma unroll
  for (int j = 0; j < 6; ++j)
    biv[j] = fused ? bi[cR0 + (j & 1) * 16 + (j >> 1) * 256 + laneLo] : 0.0f;

  float* ys = out + (size_t)BATCH * HH;  // out = [hT (B,H) | ys (B,T,H)]

  for (int t = 0; t < TT; ++t) {
    // 0) prefetch this step's gi gather and reset flags BEFORE the barrier:
    //    these global loads are independent of the LDS h tile, so their
    //    latency hides behind the barrier wait + LDS publish.
    float gir[6][8];   // j=0..3: r,z acc init ; j=4,5: i_n for the epilogue
    if (!fused) {
#pragma unroll
      for (int j = 0; j < 6; ++j) {
        int col = cR0 + (j & 1) * 16 + (j >> 1) * 256 + laneLo;
#pragma unroll
        for (int i = 0; i < 8; ++i) {
          int b = bb + i + laneHi * 8;
          gir[j][i] = gi[((size_t)b * TT + t) * H3 + col];
        }
      }
    }
    int rflag[8];
#pragma unroll
    for (int i = 0; i < 8; ++i)
      rflag[i] = resets[(size_t)(bb + i + laneHi * 8) * TT + t];

    // 1) reset-mask the carry, publish masked h as bf16 to LDS
#pragma unroll
    for (int p = 0; p < 2; ++p) {
      int col = cR0 + p * 16 + laneLo;
#pragma unroll
      for (int i = 0; i < 8; ++i) {
        if (rflag[i]) hreg[p][i] = 0.0f;
        h_bf[(i + laneHi * 8) * HH + col] = (__bf16)hreg[p][i];
      }
    }
    if (fused) {  // stage x[:,t,:] tile as bf16
      int row = threadIdx.x >> 4;
      int c0  = (threadIdx.x & 15) * 16;
      const float* xp = x + ((size_t)(bb + row) * TT + t) * HH + c0;
#pragma unroll
      for (int q = 0; q < 16; ++q) x_bf[row * HH + c0 + q] = (__bf16)xp[q];
    }
    __syncthreads();

    // 2) accumulators: j=0,1 -> r ; j=2,3 -> z (include gi) ; j=4,5 -> h_n only
    //    (kept separate: n = tanh(i_n + r*(h_n + bn)))
    v8f acc[6], accIn[2];
#pragma unroll
    for (int j = 0; j < 6; ++j)
#pragma unroll
      for (int i = 0; i < 8; ++i)
        acc[j][i] = fused ? ((j < 4) ? biv[j] : 0.0f)
                          : ((j < 4) ? gir[j][i] : 0.0f);
#pragma unroll
    for (int p = 0; p < 2; ++p)
#pragma unroll
      for (int i = 0; i < 8; ++i) accIn[p][i] = biv[4 + p];

    // 3) WMMAs over K=256, 6 N-tiles per wave, A (LDS) + Wh (L2) fragments
    //    double-buffered with sched_barrier fences to keep the pipeline.
    if (!fused) {
      v16bf abufH[2];
      v16bf bbuf[2][6];
      abufH[0] = load_a_lds(h_bf + laneLo * HH, 0, laneHi);
#pragma unroll
      for (int j = 0; j < 6; ++j)
        bbuf[0][j] = load_b(WhT, cR0 + (j & 1) * 16 + (j >> 1) * 256, 0, lane);
#pragma unroll
      for (int kt = 0; kt < 8; ++kt) {
        const int cur = kt & 1, nxt = cur ^ 1;
        if (kt < 7) {
          abufH[nxt] = load_a_lds(h_bf + laneLo * HH, kt + 1, laneHi);
#pragma unroll
          for (int j = 0; j < 6; ++j)
            bbuf[nxt][j] =
                load_b(WhT, cR0 + (j & 1) * 16 + (j >> 1) * 256, kt + 1, lane);
        }
        __builtin_amdgcn_sched_barrier(0);  // keep kt+1 loads above kt WMMAs
#pragma unroll
        for (int j = 0; j < 6; ++j)
          acc[j] = __builtin_amdgcn_wmma_f32_16x16x32_bf16(
              false, abufH[cur], false, bbuf[cur][j], (short)0, acc[j], false,
              false);
        __builtin_amdgcn_sched_barrier(0);
      }
    } else {  // fused fallback: also accumulate x@Wi (i_n separated)
#pragma unroll
      for (int kt = 0; kt < 8; ++kt) {
        v16bf aH = load_a_lds(h_bf + laneLo * HH, kt, laneHi);
        v16bf aX = load_a_lds(x_bf + laneLo * HH, kt, laneHi);
#pragma unroll
        for (int j = 0; j < 6; ++j) {
          int nCol = cR0 + (j & 1) * 16 + (j >> 1) * 256;
          v16bf bW = load_b(WhT, nCol, kt, lane);
          acc[j] = __builtin_amdgcn_wmma_f32_16x16x32_bf16(
              false, aH, false, bW, (short)0, acc[j], false, false);
          v16bf bI = load_b(WiT, nCol, kt, lane);
          if (j < 4)
            acc[j] = __builtin_amdgcn_wmma_f32_16x16x32_bf16(
                false, aX, false, bI, (short)0, acc[j], false, false);
          else
            accIn[j - 4] = __builtin_amdgcn_wmma_f32_16x16x32_bf16(
                false, aX, false, bI, (short)0, accIn[j - 4], false, false);
        }
      }
    }

    // 4) gates + carry update + ys store
#pragma unroll
    for (int p = 0; p < 2; ++p) {
      int col = cR0 + p * 16 + laneLo;
#pragma unroll
      for (int i = 0; i < 8; ++i) {
        int b = bb + i + laneHi * 8;
        float i_n = fused ? accIn[p][i] : gir[4 + p][i];
        float r  = sigmoid_f(acc[p][i]);
        float z  = sigmoid_f(acc[2 + p][i]);
        float n  = tanh_f(i_n + r * (acc[4 + p][i] + bnv[p]));
        float hN = (1.0f - z) * n + z * hreg[p][i];
        hreg[p][i] = hN;
        ys[((size_t)b * TT + t) * HH + col] = hN;
      }
    }
    __syncthreads();  // all waves done reading h_bf before next step's writes
  }

  // final carry hT -> out[0 .. B*H)
#pragma unroll
  for (int p = 0; p < 2; ++p) {
    int col = cR0 + p * 16 + laneLo;
#pragma unroll
    for (int i = 0; i < 8; ++i)
      out[(size_t)(bb + i + laneHi * 8) * HH + col] = hreg[p][i];
  }
}

// ------------------------------- launcher ----------------------------------
extern "C" void kernel_launch(void* const* d_in, const int* in_sizes, int n_in,
                              void* d_out, int out_size, void* d_ws, size_t ws_size,
                              hipStream_t stream) {
  const float* x      = (const float*)d_in[0];
  const int*   resets = (const int*)d_in[1];
  const float* Wi     = (const float*)d_in[2];
  const float* bi     = (const float*)d_in[3];
  const float* Wh     = (const float*)d_in[4];
  const float* bn     = (const float*)d_in[5];
  const float* h0     = (const float*)d_in[6];
  float* out = (float*)d_out;

  char* wsb = (char*)d_ws;
  const size_t WT_BYTES = (size_t)H3 * HH * sizeof(__bf16);  // 384 KB each
  __bf16* WiT = (__bf16*)wsb;
  __bf16* WhT = (__bf16*)(wsb + WT_BYTES);
  const size_t GI_OFF   = 2 * WT_BYTES;
  const size_t GI_BYTES = (size_t)BATCH * TT * H3 * sizeof(float);  // ~384 MB
  float* gi = (float*)(wsb + GI_OFF);
  const bool two_phase = ws_size >= GI_OFF + GI_BYTES;

  pack_weights_kernel<<<dim3(H3, 2), 256, 0, stream>>>(Wi, Wh, WiT, WhT);
  if (two_phase)
    gi_gemm_kernel<<<(BATCH * TT) / 16, 256, 0, stream>>>(x, WiT, bi, gi);
  gru_scan_kernel<<<BATCH / 16, 256, 0, stream>>>(
      x, resets, WiT, WhT, bi, bn, h0, two_phase ? gi : nullptr, out);
}
